// LinearShift_17016660427047
// MI455X (gfx1250) — matrix operand order repaired and checked by reading, not compiled, not run on hardware
//
#include <hip/hip_runtime.h>
#include <hip/hip_bf16.h>

// ---------------------------------------------------------------------------
// LinearShift forward for MI455X (gfx1250, wave32, WMMA)
//   out = round_fixed(input) @ (±2^round(shift))^T + round_fixed(bias)
// bf16 WMMA GEMM, double-buffered LDS fed by GLOBAL_LOAD_ASYNC_TO_LDS_B128
// (ASYNCcnt path), 64x64 wave tiles (16 WMMAs per 8 fragment loads).
// ---------------------------------------------------------------------------

typedef __bf16 bf16_t;
typedef __attribute__((ext_vector_type(16))) __bf16 v16bf;
typedef __attribute__((ext_vector_type(8)))  __bf16 v8bf;
typedef __attribute__((ext_vector_type(8)))  float  v8f;
typedef __attribute__((ext_vector_type(4)))  float  v4f;

#define BM 128
#define BN 256
#define BK 32
#define LDT 56   // padded LDS row stride in bf16 (112B: 16B-aligned, conflict-lean)

__device__ __forceinline__ float round_fixed(float x) {
  // floor-quantize to 2^-16 steps, clamp to signed 16-bit integer range
  float f = floorf(x * 65536.0f) * (1.0f / 65536.0f);
  return fminf(fmaxf(f, -32768.0f), 32767.0f);
}

// Async 16B copy global -> LDS (CDNA5 ASYNCcnt path; no data VGPRs).
__device__ __forceinline__ void async_cp16(unsigned lds_off, const bf16_t* g) {
  asm volatile("global_load_async_to_lds_b128 %0, %1, off"
               :: "v"(lds_off), "v"((unsigned long long)(uintptr_t)g)
               : "memory");
}
__device__ __forceinline__ void wait_async0() {
  asm volatile("s_wait_asynccnt 0" ::: "memory");
}

// ---------------------------------------------------------------------------
// Prep kernel 1: quantize fp32 input -> bf16
// ---------------------------------------------------------------------------
__global__ __launch_bounds__(256) void quantize_input_kernel(
    const float* __restrict__ x, bf16_t* __restrict__ y, int n) {
  int i = (blockIdx.x * 256 + threadIdx.x) * 8;
  if (i + 8 > n) return;
  v4f a = *(const v4f*)(x + i);
  v4f b = *(const v4f*)(x + i + 4);
  v8bf o;
#pragma unroll
  for (int j = 0; j < 4; ++j) {
    o[j]     = (bf16_t)round_fixed(a[j]);
    o[4 + j] = (bf16_t)round_fixed(b[j]);
  }
  *(v8bf*)(y + i) = o;
}

// ---------------------------------------------------------------------------
// Prep kernel 2: w = exp2(round(shift)) * sign(clip(sign,-1,1)) -> bf16 (exact)
// ---------------------------------------------------------------------------
__device__ __forceinline__ bf16_t pow2_weight(float sh, float s) {
  float e = exp2f(rintf(sh));
  float c = fminf(fmaxf(s, -1.0f), 1.0f);
  float sg = (c > 0.0f) ? 1.0f : ((c < 0.0f) ? -1.0f : 0.0f);
  return (bf16_t)(e * sg);
}

__global__ __launch_bounds__(256) void make_weight_kernel(
    const float* __restrict__ shift, const float* __restrict__ sgn,
    bf16_t* __restrict__ w, int n) {
  int i = (blockIdx.x * 256 + threadIdx.x) * 8;
  if (i + 8 > n) return;
  v4f sh0 = *(const v4f*)(shift + i);
  v4f sh1 = *(const v4f*)(shift + i + 4);
  v4f sg0 = *(const v4f*)(sgn + i);
  v4f sg1 = *(const v4f*)(sgn + i + 4);
  v8bf o;
#pragma unroll
  for (int j = 0; j < 4; ++j) {
    o[j]     = pow2_weight(sh0[j], sg0[j]);
    o[4 + j] = pow2_weight(sh1[j], sg1[j]);
  }
  *(v8bf*)(w + i) = o;
}

// ---------------------------------------------------------------------------
// GEMM: C[M,N] = A[M,K] * W[N,K]^T + round_fixed(bias)[N]
// Block: 128x256 tile, 256 threads = 8 waves (2 along M x 4 along N).
// Wave tile: 64x64 => 4x4 WMMA 16x16 f32 accumulators (16 WMMAs / K-step).
// Double-buffered LDS fed by async global->LDS copies (BK=32).
// ---------------------------------------------------------------------------
__global__ __launch_bounds__(256) void gemm_wmma_bf16_kernel(
    const bf16_t* __restrict__ A, const bf16_t* __restrict__ W,
    const float* __restrict__ bias, float* __restrict__ C,
    int M, int Nf, int K) {
  __shared__ bf16_t As[2][BM * LDT];   // 2 * 128*56*2B = 28 KB
  __shared__ bf16_t Bs[2][BN * LDT];   // 2 * 256*56*2B = 56 KB

  const int tid  = threadIdx.x;
  const int lane = tid & 31;
  const int wave = tid >> 5;   // 0..7
  const int wm   = wave >> 2;  // 0..1 : wave row within block tile (64 rows each)
  const int wn   = wave & 3;   // 0..3 : wave col within block tile (64 cols each)
  const int m0   = blockIdx.y * BM;
  const int n0   = blockIdx.x * BN;

  const unsigned asBase = (unsigned)(uintptr_t)(&As[0][0]);
  const unsigned bsBase = (unsigned)(uintptr_t)(&Bs[0][0]);

  // Stage BMxBK A-tile (512 x 16B chunks) and BNxBK B-tile (1024 x 16B chunks)
  // via async global->LDS copies. Global reads fully coalesced.
  auto stage = [&](int buf, int k0) {
#pragma unroll
    for (int it = 0; it < 2; ++it) {
      int c  = tid + it * 256;
      int r  = c >> 2;            // 0..127
      int ch = (c & 3) * 8;       // bf16 col offset: 0,8,16,24
      async_cp16(asBase + (unsigned)((buf * BM * LDT + r * LDT + ch) * 2),
                 A + (size_t)(m0 + r) * K + k0 + ch);
    }
#pragma unroll
    for (int it = 0; it < 4; ++it) {
      int c  = tid + it * 256;
      int r  = c >> 2;            // 0..255
      int ch = (c & 3) * 8;
      async_cp16(bsBase + (unsigned)((buf * BN * LDT + r * LDT + ch) * 2),
                 W + (size_t)(n0 + r) * K + k0 + ch);
    }
  };

  // A fragment (16x32 bf16): lane holds row M=lane%16,
  // K = kh+{0..7} and kh+{16..23}, kh = (lane/16)*8.
  auto load_afrag = [&](const bf16_t* rowbase) -> v16bf {
    int kh = (lane >> 4) * 8;
    union { v16bf v; v8bf h[2]; } u;
    u.h[0] = *(const v8bf*)(rowbase + kh);
    u.h[1] = *(const v8bf*)(rowbase + kh + 16);
    return u.v;
  };
  // B fragment (32x16 bf16): lane holds col N=lane%16,
  // K = kb+{0..7} and kb+{16..23}, kb = (lane/16)*8.
  auto load_bfrag = [&](const bf16_t* rowbase) -> v16bf {
    int kb = (lane >> 4) * 8;
    union { v16bf v; v8bf h[2]; } u;
    u.h[0] = *(const v8bf*)(rowbase + kb);
    u.h[1] = *(const v8bf*)(rowbase + kb + 16);
    return u.v;
  };

  v8f acc[4][4];
#pragma unroll
  for (int fm = 0; fm < 4; ++fm)
#pragma unroll
    for (int fn = 0; fn < 4; ++fn) acc[fm][fn] = (v8f)(0.0f);

  const int nk = K / BK;
  stage(0, 0);
  wait_async0();
  __syncthreads();

  const int rsel = lane & 15;
  for (int kt = 0; kt < nk; ++kt) {
    const int buf = kt & 1;
    if (kt + 1 < nk) stage(buf ^ 1, (kt + 1) * BK);

    v16bf afrag[4], bfrag[4];
#pragma unroll
    for (int fm = 0; fm < 4; ++fm)
      afrag[fm] = load_afrag(&As[buf][(wm * 64 + fm * 16 + rsel) * LDT]);
#pragma unroll
    for (int fn = 0; fn < 4; ++fn)
      bfrag[fn] = load_bfrag(&Bs[buf][(wn * 64 + fn * 16 + rsel) * LDT]);

#pragma unroll
    for (int fm = 0; fm < 4; ++fm)
#pragma unroll
      for (int fn = 0; fn < 4; ++fn)
        acc[fm][fn] = __builtin_amdgcn_wmma_f32_16x16x32_bf16(
            /*neg_a=*/false, afrag[fm], /*neg_b=*/false, bfrag[fn],
            /*c_mod=*/(short)0, acc[fm][fn],
            /*reuse_a=*/false, /*reuse_b=*/false);

    wait_async0();     // our async copies into buf^1 have landed in LDS
    __syncthreads();   // publish to all waves before next iteration reads
  }

  // Epilogue: C/D layout — lane holds col N=lane%16; VGPR i holds row
  // M = i + (lane/16)*8. Add quantized bias per column.
#pragma unroll
  for (int fn = 0; fn < 4; ++fn) {
    const int col = n0 + wn * 64 + fn * 16 + (lane & 15);
    const float qb = round_fixed(bias[col]);
#pragma unroll
    for (int fm = 0; fm < 4; ++fm) {
      const int row = m0 + wm * 64 + fm * 16 + (lane >> 4) * 8;
      float* p = C + (size_t)row * Nf + col;
#pragma unroll
      for (int i = 0; i < 8; ++i)
        p[(size_t)i * Nf] = acc[fm][fn][i] + qb;
    }
  }
}

// ---------------------------------------------------------------------------
// Launcher
// d_in: [0]=input (M x K fp32), [1]=shift (N x K fp32), [2]=sign (N x K fp32),
//       [3]=bias (N fp32).  d_out: M x N fp32.
// d_ws: [A_bf16 (M*K)][W_bf16 (N*K)]  => 64 MB for 4096^3.
// ---------------------------------------------------------------------------
extern "C" void kernel_launch(void* const* d_in, const int* in_sizes, int n_in,
                              void* d_out, int out_size, void* d_ws, size_t ws_size,
                              hipStream_t stream) {
  const float* inp   = (const float*)d_in[0];
  const float* shift = (const float*)d_in[1];
  const float* sgn   = (const float*)d_in[2];
  const float* bias  = (const float*)d_in[3];
  float* out = (float*)d_out;

  const int OUT_F = in_sizes[3];
  const int IN_F  = in_sizes[1] / OUT_F;
  const int M     = in_sizes[0] / IN_F;

  bf16_t* Aq = (bf16_t*)d_ws;
  bf16_t* Wq = Aq + (size_t)M * IN_F;

  const int nA = M * IN_F;
  const int nW = OUT_F * IN_F;
  quantize_input_kernel<<<(nA / 8 + 255) / 256, 256, 0, stream>>>(inp, Aq, nA);
  make_weight_kernel<<<(nW / 8 + 255) / 256, 256, 0, stream>>>(shift, sgn, Wq, nW);

  dim3 grid(OUT_F / BN, M / BM);
  gemm_wmma_bf16_kernel<<<grid, 256, 0, stream>>>(Aq, Wq, bias, out, M, OUT_F, IN_F);
}
